// learnValEnv_85358180041021
// MI455X (gfx1250) — matrix operand order repaired and checked by reading, compiled, not run on hardware
//
#include <hip/hip_runtime.h>

typedef __attribute__((ext_vector_type(16))) __bf16 v16bf;
typedef __attribute__((ext_vector_type(8)))  float  v8f;

#define D_ASSET 16
#define T_N     100
#define T_P1    101
#define N_B     4096
#define HID     256
#define N_ROWS  (N_B * T_N)      /* 409600 */
#define N_RB    (N_ROWS / 16)    /* 25600  */
#define DT_F    4.0e-5f
#define INV_DT  25000.0f

/* workspace dword offsets */
#define OFF_W1    0        /* 2*16*256 = 8192  bf16-pair dwords */
#define OFF_W2    8192     /* 8*16*256 = 32768 */
#define OFF_W3    40960    /* 8*256    = 2048  */
#define OFF_QN    43008    /* 16*101   = 1616  */
#define OFF_PERM  44624    /* 1616 */
#define OFF_MUI   46240    /* 16*100 = 1600 */
#define OFF_C2    47840    /* 100 */
#define OFF_DELTA 48128    /* 409600 */
#define OFF_SQE   457728   /* 4096 */
#define OFF_Y0    461824   /* 4096 */

union BFV { v16bf v; unsigned int u[8]; };

__device__ __forceinline__ unsigned short f2bf(float f) {
  unsigned int u = __float_as_uint(f);
  u += 0x7FFFu + ((u >> 16) & 1u);          /* round-to-nearest-even */
  return (unsigned short)(u >> 16);
}

/* B fragment (32x16 bf16): lane L -> N = L%16, Khalf = L/16; VGPR v holds K = Khalf*16 + 2v, +1.
   Stored pre-scrambled in ws so each lane reads 8 consecutive dwords. */
__device__ __forceinline__ v16bf ldB(const float* wsf, int off, int lane) {
  const uint4* p = (const uint4*)(wsf + off + lane * 8);
  uint4 x0 = p[0], x1 = p[1];
  BFV r;
  r.u[0] = x0.x; r.u[1] = x0.y; r.u[2] = x0.z; r.u[3] = x0.w;
  r.u[4] = x1.x; r.u[5] = x1.y; r.u[6] = x1.z; r.u[7] = x1.w;
  return r.v;
}

/* A fragment (16x32 bf16) from row-major LDS tile.
   lane L -> M = L%16, half = L/16; VGPR v holds K = half*16 + 2v, +1 (matches B fragment). */
__device__ __forceinline__ v16bf ldA(const unsigned short* base, int strideH, int kbase,
                                     int m, int hf) {
  BFV r;
#pragma unroll
  for (int vv = 0; vv < 8; ++vv) {
    int kk = (hf << 4) + (vv << 1);
    r.u[vv] = *(const unsigned int*)(base + m * strideH + kbase + kk);
  }
  return r.v;
}

#define WMMA_BF16(A, B, C) \
  __builtin_amdgcn_wmma_f32_16x16x32_bf16(false, (A), false, (B), (short)0, (C), false, false)

/* ---------------- prep: repack weights to bf16 fragment layout + small precomputes -------- */
__global__ void prepK(const float* __restrict__ q, const float* __restrict__ kappa,
                      const float* __restrict__ eta, const float* __restrict__ W1,
                      const float* __restrict__ W2, const float* __restrict__ W3,
                      float* __restrict__ ws) {
  int gid = blockIdx.x * blockDim.x + threadIdx.x;
  unsigned int* wsu = (unsigned int*)ws;
  if (gid < 8192) {                                    /* W1 frags, K padded to 64 */
    int frag = gid >> 8, rem = gid & 255, lane = rem >> 3, vv = rem & 7;
    int kt = frag >> 4, nt = frag & 15;
    int N = nt * 16 + (lane & 15), hf = lane >> 4;
    int K = kt * 32 + hf * 16 + vv * 2;
    unsigned short lo = (K     < 33) ? f2bf(W1[K * 256 + N])       : (unsigned short)0;
    unsigned short hi = (K + 1 < 33) ? f2bf(W1[(K + 1) * 256 + N]) : (unsigned short)0;
    wsu[OFF_W1 + gid] = (unsigned int)lo | ((unsigned int)hi << 16);
  } else if (gid < 40960) {                            /* W2 frags */
    int idx = gid - 8192;
    int frag = idx >> 8, rem = idx & 255, lane = rem >> 3, vv = rem & 7;
    int kt = frag >> 4, nt = frag & 15;
    int N = nt * 16 + (lane & 15), hf = lane >> 4;
    int K = kt * 32 + hf * 16 + vv * 2;
    unsigned short lo = f2bf(W2[K * 256 + N]);
    unsigned short hi = f2bf(W2[(K + 1) * 256 + N]);
    wsu[OFF_W2 + idx] = (unsigned int)lo | ((unsigned int)hi << 16);
  } else if (gid < 43008) {                            /* W3 frags (N = 16) */
    int idx = gid - 40960;
    int kt = idx >> 8, rem = idx & 255, lane = rem >> 3, vv = rem & 7;
    int N = lane & 15, hf = lane >> 4;
    int K = kt * 32 + hf * 16 + vv * 2;
    unsigned short lo = f2bf(W3[K * 16 + N]);
    unsigned short hi = f2bf(W3[(K + 1) * 16 + N]);
    wsu[OFF_W3 + idx] = (unsigned int)lo | ((unsigned int)hi << 16);
  } else if (gid < 44624) {                            /* q_norm + perm */
    int i = gid - 43008;
    int d = i / T_P1, t = i - d * T_P1;
    float q0 = q[d * T_P1], qv = q[d * T_P1 + t];
    ws[OFF_QN + i]   = qv / q0;
    ws[OFF_PERM + i] = kappa[d] * (qv - q0);
  } else if (gid < 46224) {                            /* mui */
    int i = gid - 44624;
    int d = i / T_N, t = i - d * T_N;
    ws[OFF_MUI + i] = (q[d * T_P1 + t + 1] - q[d * T_P1 + t]) * INV_DT;
  } else if (gid < 46324) {                            /* c2[t] = sum_d eta*mui^2 */
    int t = gid - 46224;
    float s = 0.f;
    for (int d = 0; d < D_ASSET; ++d) {
      float mu = (q[d * T_P1 + t + 1] - q[d * T_P1 + t]) * INV_DT;
      s += eta[d] * mu * mu;
    }
    ws[OFF_C2 + t] = s;
  }
}

/* ---------------- main: fused 3-layer MLP via WMMA + delta computation -------------------- */
__global__ void __launch_bounds__(128)
mlpK(const float* __restrict__ S, const float* __restrict__ b1,
     const float* __restrict__ b2, const float* __restrict__ b3,
     float* __restrict__ ws) {
  extern __shared__ char smem[];
  const int lane = threadIdx.x & 31;
  const int wid  = threadIdx.x >> 5;
  const int m  = lane & 15;
  const int hf = lane >> 4;
  char* wbase = smem + wid * 12288;
  unsigned short* xt = (unsigned short*)(wbase);          /* 16x64 bf16  : 2 KB */
  unsigned short* hb = (unsigned short*)(wbase + 2048);   /* 16x256 bf16 : 8 KB */
  unsigned short* st = (unsigned short*)(wbase + 10240);  /* 16x32 bf16  : 1 KB */
  float*          zt = (float*)(wbase + 11264);           /* 16x16 f32   : 1 KB */

  const float* qn  = ws + OFF_QN;
  const float* pm  = ws + OFF_PERM;
  const float* mui = ws + OFF_MUI;
  const float* c2  = ws + OFF_C2;
  float* delta = ws + OFF_DELTA;

  const int nWaves = (int)gridDim.x * 4;
  for (int rb = (int)blockIdx.x * 4 + wid; rb < N_RB; rb += nWaves) {
    const int n0 = rb * 16;

    /* ---- build X tile: [tau, q_norm(16), S - S0(16), zero-pad to 64] ---- */
    for (int r2 = 0; r2 < 16; ++r2) {
      int n = n0 + r2, bb = n / T_N, t = n - bb * T_N;
      const float* Sb = S + (size_t)bb * (D_ASSET * T_P1);
      for (int c = lane; c < 64; c += 32) {
        float v = 0.f;
        if (c == 0)       v = (float)(T_N - t);
        else if (c <= 16) v = qn[(c - 1) * T_P1 + t];
        else if (c <= 32) v = Sb[(c - 17) * T_P1 + t] - Sb[(c - 17) * T_P1];
        xt[r2 * 64 + c] = f2bf(v);
      }
    }
    __asm__ volatile("" ::: "memory");

    /* ---- layer 1: h1 = relu(X @ W1 + b1), 16x256 ---- */
    v16bf a0 = ldA(xt, 64, 0,  m, hf);
    v16bf a1 = ldA(xt, 64, 32, m, hf);
#pragma unroll
    for (int nt = 0; nt < 16; ++nt) {
      float bias = b1[nt * 16 + m];
      v8f acc;
#pragma unroll
      for (int i = 0; i < 8; ++i) acc[i] = bias;
      v16bf w0 = ldB(ws, OFF_W1 + nt * 256,        lane);
      v16bf w1 = ldB(ws, OFF_W1 + (16 + nt) * 256, lane);
      acc = WMMA_BF16(a0, w0, acc);
      acc = WMMA_BF16(a1, w1, acc);
#pragma unroll
      for (int i = 0; i < 8; ++i) {
        float x = acc[i]; x = x > 0.f ? x : 0.f;
        hb[(i + 8 * hf) * 256 + nt * 16 + m] = f2bf(x);
      }
    }
    __asm__ volatile("" ::: "memory");

    /* ---- layer 2 + fused layer 3 ---- */
    v16bf ah[8];
#pragma unroll
    for (int kt = 0; kt < 8; ++kt) ah[kt] = ldA(hb, 256, kt * 32, m, hf);

    v8f zacc;
    {
      float zb = b3[m];
#pragma unroll
      for (int i = 0; i < 8; ++i) zacc[i] = zb;
    }
#pragma unroll
    for (int kb = 0; kb < 8; ++kb) {
#pragma unroll
      for (int sub = 0; sub < 2; ++sub) {
        int nt2 = kb * 2 + sub;
        float bias = b2[nt2 * 16 + m];
        v8f acc;
#pragma unroll
        for (int i = 0; i < 8; ++i) acc[i] = bias;
#pragma unroll
        for (int kt = 0; kt < 8; ++kt) {
          v16bf bw = ldB(ws, OFF_W2 + (kt * 16 + nt2) * 256, lane);
          acc = WMMA_BF16(ah[kt], bw, acc);
        }
#pragma unroll
        for (int i = 0; i < 8; ++i) {
          float x = acc[i]; x = x > 0.f ? x : 0.f;
          st[(i + 8 * hf) * 32 + sub * 16 + m] = f2bf(x);
        }
      }
      __asm__ volatile("" ::: "memory");
      v16bf as = ldA(st, 32, 0, m, hf);
      v16bf w3 = ldB(ws, OFF_W3 + kb * 256, lane);
      zacc = WMMA_BF16(as, w3, zacc);
      __asm__ volatile("" ::: "memory");
    }

    /* ---- z = (h2@W3 + b3)/d  -> LDS row-major ---- */
#pragma unroll
    for (int i = 0; i < 8; ++i) zt[(i + 8 * hf) * 16 + m] = zacc[i] * 0.0625f;
    __asm__ volatile("" ::: "memory");

    /* ---- per-row zdot / r_bsde / delta ---- */
    if (lane < 16) {
      int n = n0 + lane, bb = n / T_N, t = n - bb * T_N;
      const float* Sb = S + (size_t)bb * (D_ASSET * T_P1);
      float zd = 0.f, sdot = 0.f;
#pragma unroll
      for (int dd = 0; dd < D_ASSET; ++dd) {
        float s_t  = Sb[dd * T_P1 + t];
        float s_t1 = Sb[dd * T_P1 + t + 1];
        float sdw  = (s_t1 - pm[dd * T_P1 + t + 1]) - (s_t - pm[dd * T_P1 + t]);
        zd   += zt[lane * 16 + dd] * sdw;
        sdot += s_t * mui[dd * T_N + t];
      }
      float rbs = -(sdot + c2[t]);
      delta[n] = rbs * DT_F - zd;
    }
    __asm__ volatile("" ::: "memory");
  }
}

/* ---------------- per-batch reverse scan: rtg, y, squared error --------------------------- */
__global__ void perBK(const float* __restrict__ r, float* __restrict__ ws) {
  int b = blockIdx.x * blockDim.x + threadIdx.x;
  if (b >= N_B) return;
  const float* rb  = r  + (size_t)b * T_P1;
  const float* del = ws + OFF_DELTA + (size_t)b * T_N;
  float tail = rb[T_N];      /* sum_{j>i} r[j], starting at i = T-1 */
  float y    = rb[T_N];      /* y_T = rtg[:, -1] = r[:, T] */
  float err  = 0.f;
  for (int i = T_N - 1; i >= 0; --i) {
    y += del[i];
    float rtg = rb[i] + tail;
    float e = rtg - y;
    err += e * e;
    tail += rb[i];
  }
  ws[OFF_SQE + b] = err * DT_F;
  ws[OFF_Y0  + b] = y;
}

/* ---------------- deterministic tree reduction -> (MLoss, y_init) ------------------------- */
__global__ void reduceK(const float* __restrict__ ws, float* __restrict__ out) {
  __shared__ float s1[256];
  __shared__ float s2[256];
  int tid = threadIdx.x;
  float a = 0.f, b = 0.f;
  for (int i = tid; i < N_B; i += 256) {
    a += ws[OFF_SQE + i];
    b += ws[OFF_Y0  + i];
  }
  s1[tid] = a; s2[tid] = b;
  __syncthreads();
  for (int s = 128; s > 0; s >>= 1) {
    if (tid < s) { s1[tid] += s1[tid + s]; s2[tid] += s2[tid + s]; }
    __syncthreads();
  }
  if (tid == 0) {
    out[0] = s1[0] * (1.0f / (float)N_B);
    out[1] = s2[0] * (1.0f / (float)N_B);
  }
}

extern "C" void kernel_launch(void* const* d_in, const int* in_sizes, int n_in,
                              void* d_out, int out_size, void* d_ws, size_t ws_size,
                              hipStream_t stream) {
  (void)in_sizes; (void)n_in; (void)out_size; (void)ws_size;
  const float* q     = (const float*)d_in[0];
  const float* S     = (const float*)d_in[1];
  const float* r     = (const float*)d_in[2];
  const float* kappa = (const float*)d_in[3];
  const float* eta   = (const float*)d_in[4];
  const float* W1    = (const float*)d_in[5];
  const float* b1    = (const float*)d_in[6];
  const float* W2    = (const float*)d_in[7];
  const float* b2    = (const float*)d_in[8];
  const float* W3    = (const float*)d_in[9];
  const float* b3    = (const float*)d_in[10];
  float* out = (float*)d_out;
  float* ws  = (float*)d_ws;

  prepK<<<182, 256, 0, stream>>>(q, kappa, eta, W1, W2, W3, ws);
  mlpK<<<640, 128, 4 * 12288, stream>>>(S, b1, b2, b3, ws);
  perBK<<<16, 256, 0, stream>>>(r, ws);
  reduceK<<<1, 256, 0, stream>>>(ws, out);
}
